// Mamba2Layer_23304492548707
// MI455X (gfx1250) — matrix operand order repaired and checked by reading, not compiled, not run on hardware
//
#include <hip/hip_runtime.h>
#include <hip/hip_bf16.h>

// ---------------------------------------------------------------------------
// Mamba2 layer forward for MI455X (gfx1250).
//  - all matmuls via v_wmma_f32_16x16x32_bf16 (wave32, 16x16x32)
//  - GEMM global->LDS staging via the Tensor Data Mover (tensor_load_to_lds,
//    s_wait_tensorcnt) per CDNA5 ISA
// ---------------------------------------------------------------------------

#define D_MODEL 2048
#define D_INNER 4096
#define D_STATE 128
#define NHEADS  64
#define HEADDIM 64
#define D_CONV  4
#define CHUNK   256
#define D_XBC   4352              // D_INNER + 2*D_STATE
#define D_PROJ  8512              // 2*D_INNER + 2*D_STATE + NHEADS
#define SEQLEN  4096
#define NCHUNK  16                // SEQLEN / CHUNK
#define DT_OFF  8448              // D_INNER + D_XBC

typedef __bf16 bf16_t;
typedef bf16_t v16bf __attribute__((ext_vector_type(16)));
typedef float  v8f   __attribute__((ext_vector_type(8)));
typedef unsigned int u32x4 __attribute__((ext_vector_type(4)));
typedef int i32x8 __attribute__((ext_vector_type(8)));
typedef int i32x4 __attribute__((ext_vector_type(4)));

// ---- bf16 helpers (truncating; 1 VALU op per element or per pair) ----------
__device__ __forceinline__ bf16_t bf16_trunc(float f) {
  unsigned int u = __builtin_bit_cast(unsigned int, f);
  return __builtin_bit_cast(bf16_t, (unsigned short)(u >> 16));
}
__device__ __forceinline__ unsigned pack_bf16x2(float lo, float hi) {
  unsigned a = __builtin_bit_cast(unsigned, lo);
  unsigned b = __builtin_bit_cast(unsigned, hi);
  return (a >> 16) | (b & 0xffff0000u);
}

__device__ __forceinline__ v8f wmma_bf16(v16bf a, v16bf b, v8f c) {
  return __builtin_amdgcn_wmma_f32_16x16x32_bf16(
      /*neg_a=*/false, a, /*neg_b=*/false, b,
      /*c_mod=*/(short)0, c, /*reuse_a=*/false, /*reuse_b=*/false);
}

// A fragment: 16x32 bf16 A-matrix, row-major [m][k] in LDS (ld = row stride)
__device__ __forceinline__ v16bf load_a_frag(const bf16_t* base, int ld, int lane) {
  int row = lane & 15;
  int khalf = (lane >> 4) << 4;
  v16bf a;
#pragma unroll
  for (int e = 0; e < 16; ++e) {
    int kk = khalf + e;
    a[e] = base[row * ld + kk];
  }
  return a;
}

// B fragment where memory holds B as [n][k] row-major (contiguous reads)
__device__ __forceinline__ v16bf load_b_frag_nk(const bf16_t* base, int ld, int lane) {
  int col = lane & 15;
  int kbase = (lane >> 4) << 4;
  v16bf b;
#pragma unroll
  for (int e = 0; e < 16; ++e) b[e] = base[col * ld + kbase + e];
  return b;
}

// B fragment where memory holds B as [k][n] row-major (strided reads)
__device__ __forceinline__ v16bf load_b_frag_kn(const bf16_t* base, int ld, int lane) {
  int col = lane & 15;
  int kbase = (lane >> 4) << 4;
  v16bf b;
#pragma unroll
  for (int e = 0; e < 16; ++e) b[e] = base[(kbase + e) * ld + col];
  return b;
}

__device__ __forceinline__ float siluf(float x) { return x / (1.f + __expf(-x)); }
__device__ __forceinline__ float softplusf(float x) {
  return (x > 20.f) ? x : log1pf(__expf(x));
}

// ---- Tensor Data Mover descriptor builders (D# groups, ISA 08 §8.3/8.4) ----
__device__ __forceinline__ u32x4 tdm_group0(unsigned lds_off,
                                            unsigned long long gaddr) {
  u32x4 g;
  g[0] = 1u;                                                  // count=1, user D#
  g[1] = lds_off;                                             // lds_addr (bytes)
  g[2] = (unsigned)(gaddr & 0xffffffffu);                     // global_addr lo
  g[3] = (unsigned)((gaddr >> 32) & 0x1ffffffu) | (2u << 30); // hi | type=2
  return g;
}
__device__ __forceinline__ i32x8 tdm_group1(unsigned tdim0, unsigned tdim1,
                                            unsigned tile0, unsigned tile1,
                                            unsigned long long stride0) {
  i32x8 g;
  g[0] = (int)(2u << 16);                                     // data_size=4B
  g[1] = (int)((tdim0 & 0xffffu) << 16);
  g[2] = (int)(((tdim0 >> 16) & 0xffffu) | ((tdim1 & 0xffffu) << 16));
  g[3] = (int)(((tdim1 >> 16) & 0xffffu) | ((tile0 & 0xffffu) << 16));
  g[4] = (int)(tile1 & 0xffffu);                              // tile_dim2=0
  g[5] = (int)(stride0 & 0xffffffffu);
  g[6] = (int)((stride0 >> 32) & 0xffffu);                    // stride1=0
  g[7] = 0;
  return g;
}

// ---------------------------------------------------------------------------
// Kernel 1/8: C[M,N] = A[M,K] @ B[N,K]^T  (fp32 in/out, bf16 WMMA)
// Block tile 128x64, 8 waves. Tiles DMAed into LDS by the TDM each k-step.
// ---------------------------------------------------------------------------
__global__ __launch_bounds__(256) void gemm_rrT_bf16_wmma(
    const float* __restrict__ A, const float* __restrict__ B,
    float* __restrict__ C, int M, int N, int K) {
  __shared__ float  Afp[128][32];   // TDM landing pads (fp32)
  __shared__ float  Bfp[64][32];
  __shared__ bf16_t As[128][32];    // packed bf16 compute tiles
  __shared__ bf16_t Bs[64][32];
  const int tid  = threadIdx.x;
  const int lane = tid & 31;
  const int wave = tid >> 5;
  const int m0 = blockIdx.y * 128;
  const int n0 = blockIdx.x * 64;
  const unsigned ldsA = (unsigned)(uintptr_t)&Afp[0][0];
  const unsigned ldsB = (unsigned)(uintptr_t)&Bfp[0][0];
  unsigned* As32 = (unsigned*)&As[0][0];
  unsigned* Bs32 = (unsigned*)&Bs[0][0];
  const i32x4 z4 = {0, 0, 0, 0};
  const i32x8 z8 = {0, 0, 0, 0, 0, 0, 0, 0};

  v8f acc[4] = {};

  for (int k0 = 0; k0 < K; k0 += 32) {
    // ---- async tensor DMA: 128x32 A tile and 64x32 B tile -> LDS (fp32) ----
    if (wave == 0) {
      unsigned long long ga =
          (unsigned long long)(uintptr_t)(A + (size_t)m0 * K + k0);
      unsigned long long gb =
          (unsigned long long)(uintptr_t)(B + (size_t)n0 * K + k0);
      u32x4 g0a = tdm_group0(ldsA, ga);
      u32x4 g0b = tdm_group0(ldsB, gb);
      i32x8 g1a = tdm_group1((unsigned)K, (unsigned)M, 32u, 128u,
                             (unsigned long long)K);
      i32x8 g1b = tdm_group1((unsigned)K, (unsigned)N, 32u, 64u,
                             (unsigned long long)K);
      __builtin_amdgcn_tensor_load_to_lds(g0a, g1a, z4, z4, z8, 0);
      __builtin_amdgcn_tensor_load_to_lds(g0b, g1b, z4, z4, z8, 0);
      __builtin_amdgcn_s_wait_tensorcnt(0);
    }
    if (k0 + 32 < K) {   // keep next A rows warm in GL2 (global_prefetch_b8)
      __builtin_prefetch(&A[(size_t)(m0 + (tid & 127)) * K + k0 + 32], 0, 1);
    }
    __syncthreads();

    // ---- fp32 -> packed bf16 in LDS ----
#pragma unroll
    for (int i = 0; i < 8; ++i) {           // A: 2048 dword pairs
      int e = i * 256 + tid, r = e >> 4, cp = e & 15;
      As32[r * 16 + cp] = pack_bf16x2(Afp[r][2 * cp], Afp[r][2 * cp + 1]);
    }
#pragma unroll
    for (int i = 0; i < 4; ++i) {           // B: 1024 dword pairs
      int e = i * 256 + tid, r = e >> 4, cp = e & 15;
      Bs32[r * 16 + cp] = pack_bf16x2(Bfp[r][2 * cp], Bfp[r][2 * cp + 1]);
    }
    __syncthreads();

    v16bf a = load_a_frag(&As[wave * 16][0], 32, lane);
#pragma unroll
    for (int nt = 0; nt < 4; ++nt) {
      v16bf b = load_b_frag_nk(&Bs[nt * 16][0], 32, lane);
      acc[nt] = wmma_bf16(a, b, acc[nt]);
    }
    __syncthreads();  // WMMA reads done before next iteration's convert/DMA
  }

  const int rbase = m0 + wave * 16 + ((lane >> 4) << 3);
  const int cidx  = lane & 15;
#pragma unroll
  for (int nt = 0; nt < 4; ++nt)
#pragma unroll
    for (int v = 0; v < 8; ++v)
      C[(size_t)(rbase + v) * N + n0 + nt * 16 + cidx] = acc[nt][v];
}

// ---------------------------------------------------------------------------
// Kernel 2/8: causal depthwise conv1d (width 4) + SiLU over the xBC slice
// ---------------------------------------------------------------------------
__global__ __launch_bounds__(256) void conv_silu_kernel(
    const float* __restrict__ zxbcdt, const float* __restrict__ conv_w,
    float* __restrict__ xconv) {
  size_t idx = (size_t)blockIdx.x * 256 + threadIdx.x;
  if (idx >= (size_t)SEQLEN * D_XBC) return;
  int ch = (int)(idx % D_XBC);
  int t  = (int)(idx / D_XBC);
  float acc = 0.f;
#pragma unroll
  for (int j = 0; j < D_CONV; ++j) {
    int tt = t - (D_CONV - 1) + j;
    if (tt >= 0)
      acc += conv_w[ch * D_CONV + j] * zxbcdt[(size_t)tt * D_PROJ + D_INNER + ch];
  }
  xconv[idx] = siluf(acc);
}

// ---------------------------------------------------------------------------
// Kernel 3/8: dt = softplus(raw + bias); per-chunk cumsum of dt*A; chunk decay
// ---------------------------------------------------------------------------
__global__ __launch_bounds__(256) void dt_prep_kernel(
    const float* __restrict__ zxbcdt, const float* __restrict__ dt_bias,
    const float* __restrict__ A_log, float* __restrict__ dtO,
    float* __restrict__ Acs, float* __restrict__ cdecay) {
  int idx = blockIdx.x * 256 + threadIdx.x;
  if (idx >= NCHUNK * NHEADS) return;
  int c = idx >> 6, h = idx & 63;
  float A = -__expf(A_log[h]);
  float bias = dt_bias[h];
  float run = 0.f;
  for (int t = 0; t < CHUNK; ++t) {
    size_t row = (size_t)(c * CHUNK + t);
    float d = softplusf(zxbcdt[row * D_PROJ + DT_OFF + h] + bias);
    dtO[row * NHEADS + h] = d;
    run += d * A;
    Acs[((size_t)c * NHEADS + h) * CHUNK + t] = run;
  }
  cdecay[c * NHEADS + h] = __expf(run);
}

// ---------------------------------------------------------------------------
// Kernel 4/8: fused intra-chunk kernel, one block per (chunk, head).
//  phase1: S = tril(C@B^T * exp(segsum))        (WMMA, K=128)
//  scale : Bb[s][n] *= exp(Acs[255]-Acs[s])     (in-place, for phase 3)
//  phase2: Y_diag = S @ xdt  (+ D*x)            (WMMA, K=256)
//  phase3: states = xdt^T(scaled B)             (WMMA, K=256)
// LDS: Cb 64K + Bb 64K + XbT 32K + Sb 128K + Acs/dec 2K  (~291 KB dynamic)
// ---------------------------------------------------------------------------
__global__ __launch_bounds__(256) void chunk_intra_kernel(
    const float* __restrict__ xconv, const float* __restrict__ dt,
    const float* __restrict__ Acs, const float* __restrict__ D_param,
    float* __restrict__ Y, float* __restrict__ states) {
  extern __shared__ char smem[];
  bf16_t* Cb   = (bf16_t*)smem;                 // [256][128]  C[l][n]
  bf16_t* Bb   = Cb + CHUNK * D_STATE;          // [256][128]  B[s][n]
  bf16_t* XbT  = Bb + CHUNK * D_STATE;          // [64][256]   xdt[p][s] (T)
  bf16_t* Sb   = XbT + HEADDIM * CHUNK;         // [256][256]  masked scores
  float*  AcsS = (float*)(Sb + CHUNK * CHUNK);  // [256]
  float*  decS = AcsS + CHUNK;                  // [256] decay-to-end

  const int tid  = threadIdx.x;
  const int lane = tid & 31;
  const int wave = tid >> 5;
  const int c = blockIdx.x;
  const int h = blockIdx.y;

  // ---- stage B, C (pair-packed), xdt transposed, Acs, decay-to-end ----
  {
    unsigned* Bb32 = (unsigned*)Bb;
    unsigned* Cb32 = (unsigned*)Cb;
    for (int i = tid; i < CHUNK * (D_STATE / 2); i += 256) {
      int l = i >> 6, np = i & 63;
      const float* src = xconv + (size_t)(c * CHUNK + l) * D_XBC + D_INNER;
      Bb32[i] = pack_bf16x2(src[2 * np], src[2 * np + 1]);
      Cb32[i] = pack_bf16x2(src[D_STATE + 2 * np], src[D_STATE + 2 * np + 1]);
    }
    for (int i = tid; i < CHUNK * HEADDIM; i += 256) {
      int l = i >> 6, p = i & 63;
      size_t row = (size_t)(c * CHUNK + l);
      float dtv = dt[row * NHEADS + h];
      XbT[p * CHUNK + l] = bf16_trunc(xconv[row * D_XBC + h * HEADDIM + p] * dtv);
    }
    size_t abase = ((size_t)c * NHEADS + h) * CHUNK;
    float a = Acs[abase + tid];
    AcsS[tid] = a;
    decS[tid] = __expf(Acs[abase + CHUNK - 1] - a);
  }
  __syncthreads();

  // ---- phase 1: S[l][s] = sum_n C[l][n]*B[s][n], causal mask + decay ----
  for (int lti = 0; lti < 2; ++lti) {
    const int lt = wave * 2 + lti;
    for (int st = 0; st < 16; ++st) {
      v8f s = {};
#pragma unroll
      for (int ks = 0; ks < 4; ++ks) {
        v16bf a = load_a_frag(&Cb[(lt * 16) * D_STATE + ks * 32], D_STATE, lane);
        v16bf b = load_b_frag_nk(&Bb[(st * 16) * D_STATE + ks * 32], D_STATE, lane);
        s = wmma_bf16(a, b, s);
      }
      const int lrow = lt * 16 + ((lane >> 4) << 3);
      const int scol = st * 16 + (lane & 15);
#pragma unroll
      for (int v = 0; v < 8; ++v) {
        int l = lrow + v;
        float val = (l >= scol) ? s[v] * __expf(AcsS[l] - AcsS[scol]) : 0.f;
        Sb[l * CHUNK + scol] = bf16_trunc(val);
      }
    }
  }
  __syncthreads();

  // ---- in-place scale B by decay-to-end (feeds phase 3) ----
  for (int i = tid; i < CHUNK * D_STATE; i += 256)
    Bb[i] = bf16_trunc((float)Bb[i] * decS[i >> 7]);

  // ---- phase 2: Y_diag[l][p] = sum_s S[l][s] * xdt[s][p]  (+ D*x) ----
  {
    v8f y[2][4] = {};
    for (int lti = 0; lti < 2; ++lti) {
      const int lt = wave * 2 + lti;
#pragma unroll
      for (int ks = 0; ks < 8; ++ks) {
        v16bf a = load_a_frag(&Sb[(lt * 16) * CHUNK + ks * 32], CHUNK, lane);
#pragma unroll
        for (int pt = 0; pt < 4; ++pt) {
          v16bf b = load_b_frag_nk(&XbT[(pt * 16) * CHUNK + ks * 32], CHUNK, lane);
          y[lti][pt] = wmma_bf16(a, b, y[lti][pt]);
        }
      }
    }
    const float Dh = D_param[h];
    for (int lti = 0; lti < 2; ++lti) {
      const int lt = wave * 2 + lti;
      const int lrow = lt * 16 + ((lane >> 4) << 3);
#pragma unroll
      for (int pt = 0; pt < 4; ++pt) {
        const int pcol = pt * 16 + (lane & 15);
#pragma unroll
        for (int v = 0; v < 8; ++v) {
          size_t row = (size_t)(c * CHUNK + lrow + v);
          float xv = xconv[row * D_XBC + h * HEADDIM + pcol];
          Y[row * D_INNER + h * HEADDIM + pcol] = y[lti][pt][v] + Dh * xv;
        }
      }
    }
  }
  __syncthreads();   // Bb fully scaled + phase-1/2 LDS traffic retired

  // ---- phase 3: states[p][n] = sum_s xdt[s][p] * (B*dec)[s][n] ----
  {
    const int mt  = wave & 3;          // p-tile (0..3)
    const int nt0 = (wave >> 2) * 4;   // n-tiles nt0..nt0+3 (of 8)
    v8f st[4] = {};
#pragma unroll
    for (int ks = 0; ks < 8; ++ks) {
      v16bf a = load_a_frag(&XbT[(mt * 16) * CHUNK + ks * 32], CHUNK, lane);
#pragma unroll
      for (int j = 0; j < 4; ++j) {
        v16bf b = load_b_frag_kn(&Bb[(ks * 32) * D_STATE + (nt0 + j) * 16],
                                 D_STATE, lane);
        st[j] = wmma_bf16(a, b, st[j]);
      }
    }
    const int prow = mt * 16 + ((lane >> 4) << 3);
#pragma unroll
    for (int j = 0; j < 4; ++j) {
      const int n = (nt0 + j) * 16 + (lane & 15);
#pragma unroll
      for (int v = 0; v < 8; ++v) {
        size_t g = (((size_t)c * NHEADS + h) * HEADDIM + prow + v) * D_STATE + n;
        states[g] = st[j][v];
      }
    }
  }
}

// ---------------------------------------------------------------------------
// Kernel 5/8: inter-chunk recurrence (16 sequential steps)
// ---------------------------------------------------------------------------
__global__ __launch_bounds__(256) void chunk_scan_kernel(
    const float* __restrict__ states, const float* __restrict__ cdecay,
    float* __restrict__ prev) {
  size_t idx = (size_t)blockIdx.x * 256 + threadIdx.x;   // over H*P*N
  if (idx >= (size_t)NHEADS * HEADDIM * D_STATE) return;
  int h = (int)(idx >> 13);                // HEADDIM*D_STATE = 8192
  size_t off = idx & 8191;
  float run = 0.f;
  for (int c = 0; c < NCHUNK; ++c) {
    size_t g = (((size_t)c * NHEADS + h) << 13) + off;
    prev[g] = run;
    run = run * cdecay[c * NHEADS + h] + states[g];
  }
}

// ---------------------------------------------------------------------------
// Kernel 6/8: Y += exp(Acs[l]) * (C @ prev^T), one block per (chunk, head)
// ---------------------------------------------------------------------------
__global__ __launch_bounds__(256) void chunk_yoff_kernel(
    const float* __restrict__ xconv, const float* __restrict__ prev,
    const float* __restrict__ Acs, float* __restrict__ Y) {
  extern __shared__ char smem[];
  bf16_t* Cb   = (bf16_t*)smem;                  // [256][128]
  bf16_t* Pb   = Cb + CHUNK * D_STATE;           // [64][128]  prev[p][n]
  float*  AcsS = (float*)(Pb + HEADDIM * D_STATE);

  const int tid  = threadIdx.x;
  const int lane = tid & 31;
  const int wave = tid >> 5;
  const int c = blockIdx.x;
  const int h = blockIdx.y;

  {
    unsigned* Cb32 = (unsigned*)Cb;
    for (int i = tid; i < CHUNK * (D_STATE / 2); i += 256) {
      int l = i >> 6, np = i & 63;
      const float* src =
          xconv + (size_t)(c * CHUNK + l) * D_XBC + D_INNER + D_STATE;
      Cb32[i] = pack_bf16x2(src[2 * np], src[2 * np + 1]);
    }
    unsigned* Pb32 = (unsigned*)Pb;
    const float* psrc = prev + (((size_t)c * NHEADS + h) * HEADDIM) * D_STATE;
    for (int i = tid; i < HEADDIM * (D_STATE / 2); i += 256)
      Pb32[i] = pack_bf16x2(psrc[2 * i], psrc[2 * i + 1]);
    AcsS[tid] = Acs[((size_t)c * NHEADS + h) * CHUNK + tid];
  }
  __syncthreads();

  v8f y[2][4] = {};
  for (int lti = 0; lti < 2; ++lti) {
    const int lt = wave * 2 + lti;
#pragma unroll
    for (int ks = 0; ks < 4; ++ks) {
      v16bf a = load_a_frag(&Cb[(lt * 16) * D_STATE + ks * 32], D_STATE, lane);
#pragma unroll
      for (int pt = 0; pt < 4; ++pt) {
        v16bf b = load_b_frag_nk(&Pb[(pt * 16) * D_STATE + ks * 32], D_STATE, lane);
        y[lti][pt] = wmma_bf16(a, b, y[lti][pt]);
      }
    }
  }
  for (int lti = 0; lti < 2; ++lti) {
    const int lt = wave * 2 + lti;
    const int lrow = lt * 16 + ((lane >> 4) << 3);
#pragma unroll
    for (int pt = 0; pt < 4; ++pt) {
      const int pcol = pt * 16 + (lane & 15);
#pragma unroll
      for (int v = 0; v < 8; ++v) {
        int l = lrow + v;
        size_t row = (size_t)(c * CHUNK + l);
        Y[row * D_INNER + h * HEADDIM + pcol] += __expf(AcsS[l]) * y[lti][pt][v];
      }
    }
  }
}

// ---------------------------------------------------------------------------
// Kernel 7/8: gated RMSNorm: yn = normalize(Y * silu(z)) * norm_w, per token
// ---------------------------------------------------------------------------
__global__ __launch_bounds__(256) void gated_rmsnorm_kernel(
    const float* __restrict__ Y, const float* __restrict__ zxbcdt,
    const float* __restrict__ norm_w, float* __restrict__ yn) {
  __shared__ float red[256];
  const int t = blockIdx.x;
  float yg[16];
  float ss = 0.f;
#pragma unroll
  for (int i = 0; i < 16; ++i) {
    int j = i * 256 + threadIdx.x;
    float z = zxbcdt[(size_t)t * D_PROJ + j];
    float v = Y[(size_t)t * D_INNER + j] * siluf(z);
    yg[i] = v;
    ss += v * v;
  }
  red[threadIdx.x] = ss;
  __syncthreads();
  for (int s = 128; s > 0; s >>= 1) {
    if (threadIdx.x < s) red[threadIdx.x] += red[threadIdx.x + s];
    __syncthreads();
  }
  const float inv = rsqrtf(red[0] / (float)D_INNER + 1e-5f);
#pragma unroll
  for (int i = 0; i < 16; ++i) {
    int j = i * 256 + threadIdx.x;
    yn[(size_t)t * D_INNER + j] = yg[i] * inv * norm_w[j];
  }
}

// ---------------------------------------------------------------------------
// Host-side orchestration
// ---------------------------------------------------------------------------
extern "C" void kernel_launch(void* const* d_in, const int* in_sizes, int n_in,
                              void* d_out, int out_size, void* d_ws, size_t ws_size,
                              hipStream_t stream) {
  const float* u         = (const float*)d_in[0];
  const float* in_proj_w = (const float*)d_in[1];
  const float* conv_w    = (const float*)d_in[2];
  const float* dt_bias   = (const float*)d_in[3];
  const float* A_log     = (const float*)d_in[4];
  const float* D_param   = (const float*)d_in[5];
  const float* norm_w    = (const float*)d_in[6];
  const float* out_proj_w= (const float*)d_in[7];
  float* out = (float*)d_out;

  char* ws = (char*)d_ws;
  size_t off = 0;
  float* zxbcdt = (float*)(ws + off); off += (size_t)SEQLEN * D_PROJ * 4;
  float* xconv  = (float*)(ws + off); off += (size_t)SEQLEN * D_XBC * 4;
  float* dtb    = (float*)(ws + off); off += (size_t)SEQLEN * NHEADS * 4;
  float* Acs    = (float*)(ws + off); off += (size_t)NCHUNK * NHEADS * CHUNK * 4;
  float* cdecay = (float*)(ws + off); off += (size_t)NCHUNK * NHEADS * 4;
  float* Y      = (float*)(ws + off); off += (size_t)SEQLEN * D_INNER * 4;
  float* states = (float*)(ws + off); off += (size_t)NCHUNK * NHEADS * HEADDIM * D_STATE * 4;
  float* prev   = (float*)(ws + off); off += (size_t)NCHUNK * NHEADS * HEADDIM * D_STATE * 4;
  float* yn     = (float*)(ws + off); off += (size_t)SEQLEN * D_INNER * 4;

  // 1) in_proj
  gemm_rrT_bf16_wmma<<<dim3(D_PROJ / 64, SEQLEN / 128), 256, 0, stream>>>(
      u, in_proj_w, zxbcdt, SEQLEN, D_PROJ, D_MODEL);

  // 2) causal depthwise conv + SiLU
  {
    size_t n = (size_t)SEQLEN * D_XBC;
    conv_silu_kernel<<<(unsigned)((n + 255) / 256), 256, 0, stream>>>(
        zxbcdt, conv_w, xconv);
  }

  // 3) dt softplus + per-chunk dA cumsum + chunk decay
  dt_prep_kernel<<<(NCHUNK * NHEADS + 255) / 256, 256, 0, stream>>>(
      zxbcdt, dt_bias, A_log, dtb, Acs, cdecay);

  // 4) fused intra-chunk
  {
    size_t smem = ((size_t)CHUNK * D_STATE * 2 + (size_t)HEADDIM * CHUNK +
                   (size_t)CHUNK * CHUNK) * sizeof(bf16_t) +
                  2 * CHUNK * sizeof(float);
    chunk_intra_kernel<<<dim3(NCHUNK, NHEADS), 256, smem, stream>>>(
        xconv, dtb, Acs, D_param, Y, states);
  }

  // 5) inter-chunk scan
  {
    size_t n = (size_t)NHEADS * HEADDIM * D_STATE;
    chunk_scan_kernel<<<(unsigned)((n + 255) / 256), 256, 0, stream>>>(
        states, cdecay, prev);
  }

  // 6) Y += exp(Acs) * (C @ prev^T)
  {
    size_t smem = ((size_t)CHUNK * D_STATE + (size_t)HEADDIM * D_STATE) *
                      sizeof(bf16_t) +
                  CHUNK * sizeof(float);
    chunk_yoff_kernel<<<dim3(NCHUNK, NHEADS), 256, smem, stream>>>(
        xconv, prev, Acs, Y);
  }

  // 7) gated RMSNorm
  gated_rmsnorm_kernel<<<SEQLEN, 256, 0, stream>>>(Y, zxbcdt, norm_w, yn);

  // 8) out_proj
  gemm_rrT_bf16_wmma<<<dim3(D_MODEL / 64, SEQLEN / 128), 256, 0, stream>>>(
      yn, out_proj_w, out, SEQLEN, D_MODEL, D_INNER);

  (void)in_sizes; (void)n_in; (void)out_size; (void)ws_size;
}
